// CosSimilarity_25563645345998
// MI455X (gfx1250) — matrix operand order repaired and checked by reading, not compiled, not run on hardware
//
#include <hip/hip_runtime.h>
#include <hip/hip_bf16.h>

typedef _Float16 v16h __attribute__((ext_vector_type(16)));
typedef _Float16 v8h  __attribute__((ext_vector_type(8)));
typedef _Float16 v4h  __attribute__((ext_vector_type(4)));
typedef float    v8f  __attribute__((ext_vector_type(8)));

#define NROWS   4096
#define DIMK    1024
#define INVTEMP 20.0f   // 1 / 0.05

// ---------------------------------------------------------------------------
// Pass 1: per-row L2-normalize f32 -> f16.  One 256-thread block per row.
// ---------------------------------------------------------------------------
__global__ __launch_bounds__(256) void rownorm_f32_to_f16(
    const float* __restrict__ in, _Float16* __restrict__ out)
{
    const int row = blockIdx.x;
    const int t   = threadIdx.x;

    const float4* src = (const float4*)(in + (size_t)row * DIMK);
    float4 v = src[t];                       // 256 threads * 4 = 1024 elems
    float  ss = v.x * v.x + v.y * v.y + v.z * v.z + v.w * v.w;

    __shared__ float red[256];
    red[t] = ss;
    __syncthreads();
    #pragma unroll
    for (int s = 128; s > 0; s >>= 1) {
        if (t < s) red[t] += red[t + s];
        __syncthreads();
    }
    const float inv = 1.0f / fmaxf(sqrtf(red[0]), 1e-8f);

    v4h o;
    o[0] = (_Float16)(v.x * inv);
    o[1] = (_Float16)(v.y * inv);
    o[2] = (_Float16)(v.z * inv);
    o[3] = (_Float16)(v.w * inv);
    *(v4h*)(out + (size_t)row * DIMK + (size_t)t * 4) = o;
}

// ---------------------------------------------------------------------------
// Pass 2: C[i][j] = 20 * dot(xh[i,:], yh[j,:]) via v_wmma_f32_16x16x32_f16.
// Block tile 256(M) x 128(N), 8 waves, wave tile 64x64 (4x4 fragments).
// BK=32 halves per stage, LDS double-buffered -> 1 barrier per K-step.
// ---------------------------------------------------------------------------
#define BM 256
#define BN 128
#define BK 32
#define NSTEP (DIMK / BK)          // 32
#define LDSTRIDE 40                // 32 halves + 8 pad (80 B rows)

__global__ __launch_bounds__(256) void cosgemm_wmma(
    const _Float16* __restrict__ X,   // [4096,1024] f16 normalized x
    const _Float16* __restrict__ Y,   // [4096,1024] f16 normalized y
    float* __restrict__ C)            // [4096,4096] f32
{
    __shared__ _Float16 sx[2][BM * LDSTRIDE];   // 2 x 20480 B
    __shared__ _Float16 sy[2][BN * LDSTRIDE];   // 2 x 10240 B  (total 60 KB)

    const int t    = threadIdx.x;
    const int wid  = t >> 5;
    const int lane = t & 31;
    const int lm   = lane & 15;
    const int hi   = lane >> 4;

    const int i0 = blockIdx.y * BM;
    const int j0 = blockIdx.x * BN;
    const int wi = (wid >> 1) * 64;   // wave M offset: 0,64,128,192
    const int wj = (wid & 1)  * 64;   // wave N offset: 0,64

    // per-thread staging coordinates (constant over K)
    int xr[4], xc[4], yr[2], yc[2];
    #pragma unroll
    for (int it = 0; it < 4; ++it) { int c = t + it * 256; xr[it] = c >> 2; xc[it] = (c & 3) * 8; }
    #pragma unroll
    for (int it = 0; it < 2; ++it) { int c = t + it * 256; yr[it] = c >> 2; yc[it] = (c & 3) * 8; }

    v8f acc[4][4] = {};
    v8h rx[4], ry[2];

    // ---- prologue: stage K-step 0 into buffer 0 ----
    #pragma unroll
    for (int it = 0; it < 4; ++it)
        rx[it] = *(const v8h*)(X + (size_t)(i0 + xr[it]) * DIMK + xc[it]);
    #pragma unroll
    for (int it = 0; it < 2; ++it)
        ry[it] = *(const v8h*)(Y + (size_t)(j0 + yr[it]) * DIMK + yc[it]);
    #pragma unroll
    for (int it = 0; it < 4; ++it)
        *(v8h*)(&sx[0][xr[it] * LDSTRIDE + xc[it]]) = rx[it];
    #pragma unroll
    for (int it = 0; it < 2; ++it)
        *(v8h*)(&sy[0][yr[it] * LDSTRIDE + yc[it]]) = ry[it];
    __syncthreads();

    int buf = 0;
    for (int ks = 0; ks < NSTEP; ++ks) {
        const int kn = (ks + 1) * BK;

        // ---- prefetch next stage into VGPRs (hides under the 16 WMMAs) ----
        if (ks + 1 < NSTEP) {
            #pragma unroll
            for (int it = 0; it < 4; ++it) {
                const _Float16* g = X + (size_t)(i0 + xr[it]) * DIMK + kn + xc[it];
                rx[it] = *(const v8h*)g;
                if (kn + BK < DIMK) __builtin_prefetch(g + BK, 0, 3);
            }
            #pragma unroll
            for (int it = 0; it < 2; ++it) {
                const _Float16* g = Y + (size_t)(j0 + yr[it]) * DIMK + kn + yc[it];
                ry[it] = *(const v8h*)g;
                if (kn + BK < DIMK) __builtin_prefetch(g + BK, 0, 3);
            }
        }

        // ---- fragment loads from current buffer (ISA VGPR layouts) ----
        v16h a[4], b[4];
        #pragma unroll
        for (int f = 0; f < 4; ++f) {
            // A 16x32 f16: lane l -> row l%16; K = {hi*8..+7, 16+hi*8..+7}
            const _Float16* p = &sx[buf][(wi + f * 16 + lm) * LDSTRIDE];
            v8h lo = *(const v8h*)(p + hi * 8);
            v8h h8 = *(const v8h*)(p + 16 + hi * 8);
            a[f] = __builtin_shufflevector(lo, h8,
                    0,1,2,3,4,5,6,7,8,9,10,11,12,13,14,15);
        }
        #pragma unroll
        for (int f = 0; f < 4; ++f) {
            // B 32x16 f16: lane l -> col l%16; K = {hi*8..+7, 16+hi*8..+7}
            const _Float16* p = &sy[buf][(wj + f * 16 + lm) * LDSTRIDE];
            v8h lo = *(const v8h*)(p + hi * 8);
            v8h h8 = *(const v8h*)(p + 16 + hi * 8);
            b[f] = __builtin_shufflevector(lo, h8,
                    0,1,2,3,4,5,6,7,8,9,10,11,12,13,14,15);
        }

        // ---- 16 WMMAs per K-step ----
        #pragma unroll
        for (int fm = 0; fm < 4; ++fm)
            #pragma unroll
            for (int fn = 0; fn < 4; ++fn)
                acc[fm][fn] = __builtin_amdgcn_wmma_f32_16x16x32_f16(
                    false, a[fm], false, b[fn],
                    (short)0, acc[fm][fn], false, false);

        // ---- store next stage into the other buffer ----
        if (ks + 1 < NSTEP) {
            #pragma unroll
            for (int it = 0; it < 4; ++it)
                *(v8h*)(&sx[buf ^ 1][xr[it] * LDSTRIDE + xc[it]]) = rx[it];
            #pragma unroll
            for (int it = 0; it < 2; ++it)
                *(v8h*)(&sy[buf ^ 1][yr[it] * LDSTRIDE + yc[it]]) = ry[it];
        }
        __syncthreads();
        buf ^= 1;
    }

    // ---- epilogue: scale by 1/TEMP, store f32 ----
    // C-frag layout: VGPR v, lane l -> M = v + 8*(l>=16), N = l%16
    #pragma unroll
    for (int fm = 0; fm < 4; ++fm) {
        #pragma unroll
        for (int fn = 0; fn < 4; ++fn) {
            #pragma unroll
            for (int v = 0; v < 8; ++v) {
                const int m = i0 + wi + fm * 16 + v + hi * 8;
                const int n = j0 + wj + fn * 16 + lm;
                C[(size_t)m * NROWS + n] = acc[fm][fn][v] * INVTEMP;
            }
        }
    }
}

// ---------------------------------------------------------------------------
extern "C" void kernel_launch(void* const* d_in, const int* in_sizes, int n_in,
                              void* d_out, int out_size, void* d_ws, size_t ws_size,
                              hipStream_t stream)
{
    (void)in_sizes; (void)n_in; (void)out_size; (void)ws_size;

    const float* x = (const float*)d_in[0];
    const float* y = (const float*)d_in[1];
    float* out     = (float*)d_out;

    _Float16* xh = (_Float16*)d_ws;                       // 8 MB
    _Float16* yh = xh + (size_t)NROWS * DIMK;             // 8 MB

    rownorm_f32_to_f16<<<NROWS, 256, 0, stream>>>(x, xh);
    rownorm_f32_to_f16<<<NROWS, 256, 0, stream>>>(y, yh);

    dim3 grid(NROWS / BN, NROWS / BM);                    // (32, 16)
    cosgemm_wmma<<<grid, 256, 0, stream>>>(xh, yh, out);
}
